// KWattentionLayer_45226005627029
// MI455X (gfx1250) — matrix operand order repaired and checked by reading, not compiled, not run on hardware
//
#include <hip/hip_runtime.h>
#include <hip/hip_bf16.h>

typedef _Float16 f16;
typedef _Float16 v16h __attribute__((ext_vector_type(16)));
typedef _Float16 v8h  __attribute__((ext_vector_type(8)));
typedef _Float16 v2h  __attribute__((ext_vector_type(2)));
typedef float    v8f  __attribute__((ext_vector_type(8)));

#define G_GROUPS 32
#define NKW 100
#define Dm 768
#define Hh 12
#define Ss 256
#define Bb 2

// ---------- WMMA helpers ----------

__device__ __forceinline__ v8f wmma16(v16h a, v16h b, v8f c) {
    return __builtin_amdgcn_wmma_f32_16x16x32_f16(false, a, false, b, (short)0, c, false, false);
}

// A-fragment (16x32 f16, M x K): lane<16: row=rowbase+lane, halves0-7=K0..7, halves8-15=K16..23
//                                lane>=16: same rows, halves0-7=K8..15, halves8-15=K24..31
__device__ __forceinline__ v16h load_a_frag(const f16* base, int rowbase, int stride, int kbase, int lane) {
    int l15 = lane & 15;
    int row = rowbase + l15;
    int o0 = kbase + ((lane < 16) ? 0 : 8);
    int o1 = kbase + ((lane < 16) ? 16 : 24);
    const f16* rp = base + row * stride;
    v8h lo = *(const v8h*)(rp + o0);
    v8h hi = *(const v8h*)(rp + o1);
    v16h a;
#pragma unroll
    for (int i = 0; i < 8; ++i) { a[i] = lo[i]; a[i + 8] = hi[i]; }
    return a;
}

// B-fragment (32x16 f16, K x N) from contraction-major tile: tile[row=n][halves along k]
// lane<16: col=colbase+lane, halves0-7=K0..7, halves8-15=K16..23 ; lane>=16: same cols, halves0-7=K8..15, halves8-15=K24..31
__device__ __forceinline__ v16h load_b_frag(const f16* base, int colbase, int stride, int kbase, int lane) {
    int row = colbase + (lane & 15);
    int o0 = kbase + ((lane < 16) ? 0 : 8);
    int o1 = kbase + ((lane < 16) ? 16 : 24);
    const f16* rp = base + row * stride;
    v8h lo = *(const v8h*)(rp + o0);
    v8h hi = *(const v8h*)(rp + o1);
    v16h b;
#pragma unroll
    for (int i = 0; i < 8; ++i) { b[i] = lo[i]; b[i + 8] = hi[i]; }
    return b;
}

// order-preserving float <-> uint map for deterministic max via ds_max_u32
__device__ __forceinline__ unsigned fmono(float f) {
    unsigned u = __float_as_uint(f);
    return (u & 0x80000000u) ? ~u : (u | 0x80000000u);
}
__device__ __forceinline__ float funmono(unsigned u) {
    return __uint_as_float((u & 0x80000000u) ? (u & 0x7FFFFFFFu) : ~u);
}

// ---------- Kernel 0: precision prep ----------

__global__ void prep_kernel(const float* __restrict__ x, const float* __restrict__ pos,
                            const float* __restrict__ neg,
                            const float* __restrict__ Wq, const float* __restrict__ Wk,
                            const float* __restrict__ Wv, const float* __restrict__ Wo,
                            const float* __restrict__ wmlp,
                            f16* __restrict__ xh, f16* __restrict__ kwh,
                            f16* __restrict__ wqh, f16* __restrict__ wkh,
                            f16* __restrict__ wvh, f16* __restrict__ woh,
                            float* __restrict__ sumw) {
    int idx = blockIdx.x * 256 + threadIdx.x;
    if (idx < Dm * Dm) {
        wqh[idx] = (f16)Wq[idx];
        wkh[idx] = (f16)Wk[idx];
        wvh[idx] = (f16)Wv[idx];
        woh[idx] = (f16)Wo[idx];
    }
    if (idx < Bb * Ss * Dm) xh[idx] = (f16)x[idx];
    if (idx < NKW * Dm) {
        int n = idx / Dm, d = idx - n * Dm;
        float v = (n & 1) ? neg[(n >> 1) * Dm + d] : pos[(n >> 1) * Dm + d];
        kwh[idx] = (f16)v;
    }
    if (idx == 0) {
        float s = 0.f;
        for (int i = 0; i < NKW; ++i) s += wmlp[i];
        *sumw = s;
    }
}

// ---------- Kernel 1: fused keyword-bilinear attention ----------
// grid: (G_GROUPS, B, H), block: 256 (8 wave32 waves)

__launch_bounds__(256, 1)
__global__ void kw_attn_kernel(const f16* __restrict__ xh, const f16* __restrict__ kwh,
                               const f16* __restrict__ Wqh, const f16* __restrict__ Wkh,
                               const f16* __restrict__ Wvh,
                               const float* __restrict__ bq, const float* __restrict__ bk,
                               const float* __restrict__ bv,
                               const float* __restrict__ mask, const float* __restrict__ wmlp,
                               float* __restrict__ partial) {
    // LDS: union buffer (phase-A A-chunk + W^T chunks | Q-stage | weight block), K/V slabs, softmax scratch
    __shared__ __attribute__((aligned(16))) f16 sUnion[256 * 72];   // 36864 B
    __shared__ __attribute__((aligned(16))) f16 sK[64 * 72];        // 9216 B  (slab rows k-local, halves=e)
    __shared__ __attribute__((aligned(16))) f16 sVt[64 * 72];       // 9216 B  (rows e, halves=k-local)
    __shared__ unsigned sColMax[64];
    __shared__ float    sRed[16][64];                               // deterministic column-sum partials

    const int tid  = threadIdx.x;
    const int lane = tid & 31;
    const int wv   = tid >> 5;
    const int l15  = lane & 15;
    const int hi8  = (lane >> 4) << 3;   // 0 or 8: row offset of lane-half in C/D tiles
    const int g = blockIdx.x, b = blockIdx.y, h = blockIdx.z;

    f16* Achunk = sUnion;              // 256 x (32 +pad8)  stride 40
    f16* Wt0    = sUnion + 256 * 40;   // 3 x [64 x 40]
    f16* Qstage = sUnion;              // 256 x (64 +pad8)  stride 72 (reuse)
    f16* wblock = sUnion;              // 256 x 72 (reuse)

    v8f ctxacc[2][4];
#pragma unroll
    for (int i = 0; i < 2; ++i)
#pragma unroll
        for (int j = 0; j < 4; ++j) ctxacc[i][j] = (v8f)0.0f;

    for (int n = g; n < NKW; n += G_GROUPS) {
        // ---------- Phase A: Q/K/V head-slice projections ----------
        v8f qa[2][4], ka[2][4], va[2][4];
#pragma unroll
        for (int i = 0; i < 2; ++i)
#pragma unroll
            for (int j = 0; j < 4; ++j) { qa[i][j] = (v8f)0.0f; ka[i][j] = (v8f)0.0f; va[i][j] = (v8f)0.0f; }

        for (int ks = 0; ks < 24; ++ks) {
            __syncthreads();
            const int kc = ks * 32;
            // A chunk: (x ⊙ kw_n)[:, kc:kc+32] -> LDS f16 (row-major, stride 40)
            for (int idx = tid; idx < 4096; idx += 256) {
                int row = idx >> 4, cp = (idx & 15) << 1;
                v2h xv = *(const v2h*)(xh + ((size_t)(b * Ss + row)) * Dm + kc + cp);
                v2h kv = *(const v2h*)(kwh + (size_t)n * Dm + kc + cp);
                *(v2h*)(Achunk + row * 40 + cp) = xv * kv;
            }
            // W chunks transposed (rows = out-col e, halves along d), stride 40
            for (int idx = tid; idx < 1024; idx += 256) {
                int k = idx >> 5, c2 = (idx & 31) << 1;
                size_t go = (size_t)(kc + k) * Dm + h * 64 + c2;
                v2h wq2 = *(const v2h*)(Wqh + go);
                v2h wk2 = *(const v2h*)(Wkh + go);
                v2h wv2 = *(const v2h*)(Wvh + go);
                f16* t0 = Wt0;
                f16* t1 = Wt0 + 2560;
                f16* t2 = Wt0 + 5120;
                t0[c2 * 40 + k] = wq2[0]; t0[(c2 + 1) * 40 + k] = wq2[1];
                t1[c2 * 40 + k] = wk2[0]; t1[(c2 + 1) * 40 + k] = wk2[1];
                t2[c2 * 40 + k] = wv2[0]; t2[(c2 + 1) * 40 + k] = wv2[1];
            }
            __syncthreads();
            v16h af0 = load_a_frag(Achunk, 32 * wv, 40, 0, lane);
            v16h af1 = load_a_frag(Achunk, 32 * wv + 16, 40, 0, lane);
#pragma unroll
            for (int j = 0; j < 4; ++j) {
                v16h bq_ = load_b_frag(Wt0, 16 * j, 40, 0, lane);
                qa[0][j] = wmma16(af0, bq_, qa[0][j]);
                qa[1][j] = wmma16(af1, bq_, qa[1][j]);
                v16h bk_ = load_b_frag(Wt0 + 2560, 16 * j, 40, 0, lane);
                ka[0][j] = wmma16(af0, bk_, ka[0][j]);
                ka[1][j] = wmma16(af1, bk_, ka[1][j]);
                v16h bv_ = load_b_frag(Wt0 + 5120, 16 * j, 40, 0, lane);
                va[0][j] = wmma16(af0, bv_, va[0][j]);
                va[1][j] = wmma16(af1, bv_, va[1][j]);
            }
        }
        __syncthreads();   // phase-A LDS reads done before Q-stage overwrite

        // biases
#pragma unroll
        for (int j = 0; j < 4; ++j) {
            float bqv = bq[h * 64 + 16 * j + l15];
            float bkv = bk[h * 64 + 16 * j + l15];
            float bvv = bv[h * 64 + 16 * j + l15];
#pragma unroll
            for (int i = 0; i < 2; ++i)
#pragma unroll
                for (int r = 0; r < 8; ++r) { qa[i][j][r] += bqv; ka[i][j][r] += bkv; va[i][j][r] += bvv; }
        }
        // stage Q (row-major q x 64, stride 72) for A-fragment re-layout
#pragma unroll
        for (int i = 0; i < 2; ++i)
#pragma unroll
            for (int j = 0; j < 4; ++j) {
                int col = 16 * j + l15;
#pragma unroll
                for (int r = 0; r < 8; ++r) {
                    int row = 32 * wv + 16 * i + r + hi8;
                    Qstage[row * 72 + col] = (f16)qa[i][j][r];
                }
            }
        __syncthreads();
        v16h qf[2][2];
#pragma unroll
        for (int i = 0; i < 2; ++i)
#pragma unroll
            for (int t = 0; t < 2; ++t)
                qf[i][t] = load_a_frag(Qstage, 32 * wv + 16 * i, 72, 32 * t, lane);

        const float wn = wmlp[n];

        // ---------- Phase B: column-softmax attention in 64-column slabs ----------
        for (int kc2i = 0; kc2i < 4; ++kc2i) {
            const int kc2 = kc2i * 64;
            __syncthreads();   // protects wblock/slab/scratch reuse (also Q-frag reads, prior ctx reads)
            if (tid < 64) sColMax[tid] = 0u;
            if ((wv >> 1) == kc2i) {   // wave-uniform: slab rows owned by waves 2*kc2i, 2*kc2i+1
#pragma unroll
                for (int i = 0; i < 2; ++i)
#pragma unroll
                    for (int j = 0; j < 4; ++j) {
                        int col = 16 * j + l15;
#pragma unroll
                        for (int r = 0; r < 8; ++r) {
                            int grow = 32 * wv + 16 * i + r + hi8;
                            int kl = grow - kc2;
                            sK[kl * 72 + col]  = (f16)ka[i][j][r];
                            sVt[col * 72 + kl] = (f16)va[i][j][r];
                        }
                    }
            }
            __syncthreads();

            // scores = Q K^T (q x 64 slab)
            v8f sc[2][4];
#pragma unroll
            for (int i = 0; i < 2; ++i)
#pragma unroll
                for (int j = 0; j < 4; ++j) sc[i][j] = (v8f)0.0f;
#pragma unroll
            for (int j = 0; j < 4; ++j)
#pragma unroll
                for (int t = 0; t < 2; ++t) {
                    v16h bK = load_b_frag(sK, 16 * j, 72, 32 * t, lane);
                    sc[0][j] = wmma16(qf[0][t], bK, sc[0][j]);
                    sc[1][j] = wmma16(qf[1][t], bK, sc[1][j]);
                }

            // scale + mask, per-column max (deterministic via mapped-u32 ds_max)
#pragma unroll
            for (int j = 0; j < 4; ++j) {
                int kcol = kc2 + 16 * j + l15;
                float mv = mask[b * Ss + kcol];
                float lmax = -3.0e38f;
#pragma unroll
                for (int i = 0; i < 2; ++i)
#pragma unroll
                    for (int r = 0; r < 8; ++r) {
                        float s = sc[i][j][r] * 0.125f + mv;
                        sc[i][j][r] = s;
                        lmax = fmaxf(lmax, s);
                    }
                atomicMax(&sColMax[16 * j + l15], fmono(lmax));
            }
            __syncthreads();

            // exp, deterministic column sums via 16-slot partial array
            const int slot = (wv << 1) | (lane >> 4);
#pragma unroll
            for (int j = 0; j < 4; ++j) {
                float m = funmono(sColMax[16 * j + l15]);
                float ls = 0.f;
#pragma unroll
                for (int i = 0; i < 2; ++i)
#pragma unroll
                    for (int r = 0; r < 8; ++r) {
                        float e = __expf(sc[i][j][r] - m);
                        sc[i][j][r] = e;
                        ls += e;
                    }
                sRed[slot][16 * j + l15] = ls;
            }
            __syncthreads();

            // weights (pre-scaled by w_mlp[n]) -> wblock (q x k-local, stride 72)
#pragma unroll
            for (int j = 0; j < 4; ++j) {
                int col = 16 * j + l15;
                float tot = 0.f;
#pragma unroll
                for (int p = 0; p < 16; ++p) tot += sRed[p][col];
                float inv = wn / tot;
#pragma unroll
                for (int i = 0; i < 2; ++i)
#pragma unroll
                    for (int r = 0; r < 8; ++r) {
                        int row = 32 * wv + 16 * i + r + hi8;
                        wblock[row * 72 + col] = (f16)(sc[i][j][r] * inv);
                    }
            }
            __syncthreads();

            // ctx += weights @ V (contraction over k-local 64)
#pragma unroll
            for (int t = 0; t < 2; ++t) {
                v16h wf0 = load_a_frag(wblock, 32 * wv, 72, 32 * t, lane);
                v16h wf1 = load_a_frag(wblock, 32 * wv + 16, 72, 32 * t, lane);
#pragma unroll
                for (int j2 = 0; j2 < 4; ++j2) {
                    v16h bV = load_b_frag(sVt, 16 * j2, 72, 32 * t, lane);
                    ctxacc[0][j2] = wmma16(wf0, bV, ctxacc[0][j2]);
                    ctxacc[1][j2] = wmma16(wf1, bV, ctxacc[1][j2]);
                }
            }
        }
    }

    // store group-partial Σ_n w_mlp[n]*ctx_n for this (g,b,h)
#pragma unroll
    for (int i = 0; i < 2; ++i)
#pragma unroll
        for (int j = 0; j < 4; ++j) {
            int col = h * 64 + 16 * j + l15;
#pragma unroll
            for (int r = 0; r < 8; ++r) {
                int row = 32 * wv + 16 * i + r + hi8;
                partial[(((size_t)g * Bb + b) * Ss + row) * Dm + col] = ctxacc[i][j][r];
            }
        }
}

// ---------- Kernel 2: reduce group partials (fixed order -> deterministic) ----------

__global__ void reduce_kernel(const float* __restrict__ partial, f16* __restrict__ fctx) {
    int idx = blockIdx.x * 256 + threadIdx.x;
    if (idx >= Bb * Ss * Dm) return;
    float s = 0.f;
#pragma unroll 4
    for (int g = 0; g < G_GROUPS; ++g) s += partial[(size_t)g * (Bb * Ss * Dm) + idx];
    fctx[idx] = (f16)s;
}

// ---------- Kernel 3: output projection (512x768 @ 768x768) ----------
// grid (8, 12), 64x64 tile per WG

__launch_bounds__(256, 1)
__global__ void out_gemm_kernel(const f16* __restrict__ fctx, const f16* __restrict__ woh,
                                const float* __restrict__ bo, const float* __restrict__ bmlp,
                                const float* __restrict__ sumw, float* __restrict__ out) {
    __shared__ __attribute__((aligned(16))) f16 sA[64 * 40];
    __shared__ __attribute__((aligned(16))) f16 sB[64 * 40];
    const int tid = threadIdx.x, lane = tid & 31, wv = tid >> 5, l15 = lane & 15;
    const int hi8 = (lane >> 4) << 3;
    const int mb = blockIdx.x * 64, nb = blockIdx.y * 64;
    const int mt = wv >> 1, nt0 = (wv & 1) * 2;

    v8f acc[2];
    acc[0] = (v8f)0.0f; acc[1] = (v8f)0.0f;

    for (int ks = 0; ks < 24; ++ks) {
        __syncthreads();
        const int kc = ks * 32;
        for (int idx = tid; idx < 1024; idx += 256) {
            int r = idx >> 4, cp = (idx & 15) << 1;
            *(v2h*)(sA + r * 40 + cp) = *(const v2h*)(fctx + (size_t)(mb + r) * Dm + kc + cp);
        }
        for (int idx = tid; idx < 1024; idx += 256) {
            int k = idx >> 5, c2 = (idx & 31) << 1;
            v2h w2 = *(const v2h*)(woh + (size_t)(kc + k) * Dm + nb + c2);
            sB[c2 * 40 + k] = w2[0];
            sB[(c2 + 1) * 40 + k] = w2[1];
        }
        __syncthreads();
        v16h af = load_a_frag(sA, 16 * mt, 40, 0, lane);
#pragma unroll
        for (int q = 0; q < 2; ++q) {
            v16h bf = load_b_frag(sB, 16 * (nt0 + q), 40, 0, lane);
            acc[q] = wmma16(af, bf, acc[q]);
        }
    }
    float sw = *sumw, bm = *bmlp;
#pragma unroll
    for (int q = 0; q < 2; ++q) {
        int col = nb + 16 * (nt0 + q) + l15;
        float addv = sw * bo[col] + bm;
#pragma unroll
        for (int r = 0; r < 8; ++r) {
            int row = mb + 16 * mt + r + hi8;
            out[(size_t)row * Dm + col] = acc[q][r] + addv;
        }
    }
}

// ---------- host launcher ----------

extern "C" void kernel_launch(void* const* d_in, const int* in_sizes, int n_in,
                              void* d_out, int out_size, void* d_ws, size_t ws_size,
                              hipStream_t stream) {
    (void)in_sizes; (void)n_in; (void)out_size; (void)ws_size;
    const float* x    = (const float*)d_in[0];
    const float* pos  = (const float*)d_in[1];
    const float* neg  = (const float*)d_in[2];
    const float* mask = (const float*)d_in[3];
    const float* Wq   = (const float*)d_in[4];
    const float* bq   = (const float*)d_in[5];
    const float* Wk   = (const float*)d_in[6];
    const float* bk   = (const float*)d_in[7];
    const float* Wv   = (const float*)d_in[8];
    const float* bv   = (const float*)d_in[9];
    const float* Wo   = (const float*)d_in[10];
    const float* bo   = (const float*)d_in[11];
    const float* wmlp = (const float*)d_in[12];
    const float* bmlp = (const float*)d_in[13];

    char* ws = (char*)d_ws;
    f16*   xh      = (f16*)(ws + 0);            // 786432
    f16*   kwh     = (f16*)(ws + 786432);       // 153600
    f16*   wqh     = (f16*)(ws + 940032);       // 1179648
    f16*   wkh     = (f16*)(ws + 2119680);      // 1179648
    f16*   wvh     = (f16*)(ws + 3299328);      // 1179648
    f16*   woh     = (f16*)(ws + 4478976);      // 1179648
    float* sumw    = (float*)(ws + 5658624);    // 256
    f16*   fctx    = (f16*)(ws + 5658880);      // 786432
    float* partial = (float*)(ws + 6445312);    // 32 * 393216 * 4 = 50331648
    float* out     = (float*)d_out;

    prep_kernel<<<2304, 256, 0, stream>>>(x, pos, neg, Wq, Wk, Wv, Wo, wmlp,
                                          xh, kwh, wqh, wkh, wvh, woh, sumw);
    kw_attn_kernel<<<dim3(G_GROUPS, Bb, Hh), 256, 0, stream>>>(xh, kwh, wqh, wkh, wvh,
                                                               bq, bk, bv, mask, wmlp, partial);
    reduce_kernel<<<1536, 256, 0, stream>>>(partial, fctx);
    out_gemm_kernel<<<dim3(8, 12), 256, 0, stream>>>(fctx, woh, bo, bmlp, sumw, out);
}
